// DriftingLoss_65111704207365
// MI455X (gfx1250) — matrix operand order repaired and checked by reading, not compiled, not run on hardware
//
#include <hip/hip_runtime.h>

// ---------------------------------------------------------------------------
// CDNA5 (gfx1250) drifting-field loss.
// Heavy GEMMs (implicit-GEMM convs, Gram matrices, drift GEMMs) run on
// v_wmma_f32_16x16x32_f16 (f16 in, f32 accumulate), wave32.
// Conv inputs are staged per-image into LDS (via the Tensor Data Mover when
// the toolchain exposes it, else a cooperative 128-bit copy); im2col gathers
// then hit LDS instead of scattered global loads.
// ---------------------------------------------------------------------------

typedef _Float16 half_t;
typedef __attribute__((ext_vector_type(16))) _Float16 v16h;
typedef __attribute__((ext_vector_type(8)))  _Float16 v8h;
typedef __attribute__((ext_vector_type(8)))  float    v8f;
typedef __attribute__((ext_vector_type(4)))  unsigned int u32x4;
typedef __attribute__((ext_vector_type(8)))  int      i32x8;
typedef __attribute__((ext_vector_type(4)))  int      i32x4;

__device__ __forceinline__ v8f wmma_f16(v16h a, v16h b, v8f c) {
  // (neg_a, A, neg_b, B, c_mod, C, reuse_a, reuse_b)
  return __builtin_amdgcn_wmma_f32_16x16x32_f16(false, a, false, b, (short)0, c,
                                                false, false);
}

// Fragment layouts (wave32), lane: nloc = lane&15, hi = lane>>4.
//  A (16x32 f16): a[j] = A[nloc][hi*16+j]
//  B (32x16 f16): b[j] = B[hi*16+j][nloc]
//  C (16x16 f32): c[r] = C[r+hi*8][nloc]

__device__ __forceinline__ v16h load16(const half_t* p) {          // 2x b128
  v8h lo = *(const v8h*)p;
  v8h hi = *(const v8h*)(p + 8);
  return __builtin_shufflevector(lo, hi, 0, 1, 2, 3, 4, 5, 6, 7,
                                 8, 9, 10, 11, 12, 13, 14, 15);
}
__device__ __forceinline__ v16h load8x2(const half_t* p0, const half_t* p1) {
  v8h lo = *(const v8h*)p0;
  v8h hi = *(const v8h*)p1;
  return __builtin_shufflevector(lo, hi, 0, 1, 2, 3, 4, 5, 6, 7,
                                 8, 9, 10, 11, 12, 13, 14, 15);
}
__device__ __forceinline__ v16h load_afrag(const half_t* row, int kb, int hi) {
  return load16(row + kb + hi * 16);
}
__device__ __forceinline__ v16h load_bfragT(const half_t* colRow, int kb, int hi) {
  return load16(colRow + kb + hi * 16);
}

__device__ __forceinline__ float block_reduce_sum(float v, float* red) {
  int t = threadIdx.x;
  red[t] = v; __syncthreads();
  for (int s = 128; s > 0; s >>= 1) {
    if (t < s) red[t] += red[t + s];
    __syncthreads();
  }
  float r = red[0]; __syncthreads();
  return r;
}

// ------------------------------ utility kernels ----------------------------

__global__ void zero_f32_kernel(float* __restrict__ p, int n) {
  int i = blockIdx.x * 256 + threadIdx.x;
  if (i < n) p[i] = 0.f;
}

__global__ void cvt_f16_kernel(half_t* __restrict__ dst, const float* __restrict__ src, int n) {
  int i = blockIdx.x * 256 + threadIdx.x;
  if (i < n) dst[i] = (half_t)src[i];
}

// Flatten OIHW f32 weights into im2col rows (Cout x Kpad, f16, zero padded)
// and build the k -> (cin*Hi*Hi)<<4 | ky<<2 | kx gather table (-1 = pad).
__global__ void prep_weights_kernel(half_t* __restrict__ dst, int* __restrict__ tab,
                                    const float* __restrict__ src,
                                    int Cout, int Kr, int Kpad, int Hi) {
  int idx = blockIdx.x * 256 + threadIdx.x;
  if (idx < Kpad) {
    int e = -1;
    if (idx < Kr) {
      int cin = idx / 9;
      int r   = idx - cin * 9;
      int ky  = r / 3;
      int kx  = r - ky * 3;
      e = ((cin * Hi * Hi) << 4) | (ky << 2) | kx;
    }
    tab[idx] = e;
  }
  if (idx >= Cout * Kpad) return;
  int co = idx / Kpad;
  int k  = idx - co * Kpad;
  dst[idx] = (k < Kr) ? (half_t)src[(size_t)co * Kr + k] : (half_t)0.f;
}

// --------------------------- conv (implicit GEMM) --------------------------
// One block (8 waves) per image. The input image (<= 32KB f16) is staged into
// LDS once -- via the Tensor Data Mover when available -- then each wave
// computes 16x16 (Cout x spatial) tiles, gathering im2col B-fragments from
// LDS. Fuses bias + SiLU + f16 store + pooled-mean atomic accumulation.
__global__ __launch_bounds__(256) void conv_wmma_kernel(
    const half_t* __restrict__ actIn, half_t* __restrict__ actOut,
    const half_t* __restrict__ w16, const int* __restrict__ tab,
    const float* __restrict__ bias, float* __restrict__ pooled,
    int Hi, int Cout, int Ho, int CinHH, int Kpad, float invS,
    int imgBase, int colofs, int mtiles, int ntiles) {
  __shared__ half_t smem[16384];                  // up to 32KB input image

  const int img = blockIdx.x;

#if __has_builtin(__builtin_amdgcn_tensor_load_to_lds)
  // ---- async stage via TDM: 1 row of (CinHH*2/8) 8-byte elements ----
  if (threadIdx.x < 32) {
    unsigned long long ga = (unsigned long long)(uintptr_t)(actIn + (size_t)img * CinHH);
    unsigned int la = (unsigned int)(uintptr_t)smem;   // low 32 bits = LDS offset
    unsigned int n8 = (unsigned int)(CinHH >> 2);      // (# halves)*2 / 8
    u32x4 g0;
    g0[0] = 1u;                                        // count=1 (valid user D#)
    g0[1] = la;                                        // lds_addr
    g0[2] = (unsigned int)ga;                          // global_addr[31:0]
    g0[3] = (unsigned int)((ga >> 32) & 0x01FFFFFFu) | (2u << 30);  // addr[56:32]|type=2
    i32x8 g1;
    g1[0] = (int)(3u << 16);                           // data_size = 8B, mask=0
    g1[1] = (int)((n8 & 0xFFFFu) << 16);               // tensor_dim0[15:0]
    g1[2] = (int)(((n8 >> 16) & 0xFFFFu) | (1u << 16));// tensor_dim0[31:16], dim1=1
    g1[3] = (int)((n8 & 0xFFFFu) << 16);               // tile_dim0
    g1[4] = 1;                                         // tile_dim1=1, tile_dim2=0
    g1[5] = (int)n8;                                   // tensor_dim0_stride
    g1[6] = (int)((n8 & 0xFFFFu) << 16);               // tensor_dim1_stride[15:0]
    g1[7] = 0;
    i32x4 z4 = {0, 0, 0, 0};
#if defined(__clang_major__) && (__clang_major__ >= 23)
    i32x8 z8 = {};
    __builtin_amdgcn_tensor_load_to_lds(g0, g1, z4, z4, z8, 0);
#else
    __builtin_amdgcn_tensor_load_to_lds(g0, g1, z4, z4, 0);
#endif
    __builtin_amdgcn_s_wait_tensorcnt(0);
  }
#else
  // ---- fallback: cooperative 128-bit synchronous copy ----
  {
    const v8h* src = (const v8h*)(actIn + (size_t)img * CinHH);
    v8h* dst = (v8h*)smem;
    int nv = CinHH >> 3;
    for (int i = threadIdx.x; i < nv; i += 256) dst[i] = src[i];
  }
#endif
  __syncthreads();

  const int wid  = threadIdx.x >> 5;
  const int lane = threadIdx.x & 31;
  const int nloc = lane & 15;
  const int hi   = lane >> 4;
  const int N    = Ho * Ho;

  for (int tile = wid; tile < mtiles * ntiles; tile += 8) {
    const int mt = tile / ntiles;
    const int nt = tile - mt * ntiles;
    const int ng = nt * 16 + nloc;
    const bool nvalid = ng < N;
    const int oy = ng / Ho, ox = ng % Ho;         // SAME pad: lo=0, hi=1
    const int ibase = oy * 2 * Hi + ox * 2;

    const half_t* aRow = w16 + (size_t)(mt * 16 + nloc) * Kpad;
    v8f acc = {};
    for (int kb = 0; kb < Kpad; kb += 32) {
      __builtin_prefetch(aRow + kb + 64, 0, 0);
      v16h a = load_afrag(aRow, kb, hi);
      v16h b;
#pragma unroll
      for (int j = 0; j < 16; ++j) {
        int e = tab[kb + hi * 16 + j];
        half_t v = (half_t)0.f;
        if (nvalid && e >= 0) {
          int ky = (e >> 2) & 3;
          int kx = e & 3;
          if ((oy * 2 + ky) < Hi && (ox * 2 + kx) < Hi)
            v = smem[(e >> 4) + ibase + ky * Hi + kx];
        }
        b[j] = v;
      }
      acc = wmma_f16(a, b, acc);
    }

    half_t* outImg = actOut + (size_t)img * Cout * N;
    float*  prow   = pooled + (size_t)(imgBase + img) * 960 + colofs;
#pragma unroll
    for (int r = 0; r < 8; ++r) {
      int m   = mt * 16 + r + hi * 8;
      float x = acc[r] + bias[m];
      float y = x / (1.f + __expf(-x));           // SiLU
      if (nvalid) outImg[(size_t)m * N + ng] = (half_t)y;
      float s = nvalid ? y : 0.f;
#pragma unroll
      for (int off = 1; off < 16; off <<= 1) s += __shfl_xor(s, off, 16);
      if (nloc == 0) atomicAdd(&prow[m], s * invS);
    }
  }
}

// -------------------- pooled-feature normalize (per row) -------------------
__global__ void norm_pool_kernel(const float* __restrict__ pooled, int colofs, int C,
                                 half_t* __restrict__ f16, half_t* __restrict__ f16T,
                                 float* __restrict__ qf, float* __restrict__ sqn) {
  __shared__ float red[256];
  __shared__ float sc;
  const int row = blockIdx.x;
  const float* f = pooled + (size_t)row * 960 + colofs;
  float partial = 0.f;
  for (int c = threadIdx.x; c < C; c += 256) { float v = f[c]; partial += v * v; }
  float ss = block_reduce_sum(partial, red);
  if (threadIdx.x == 0) {
    float n = sqrtf(ss);
    float scale = sqrtf((float)C) / fmaxf(n, 1e-12f);
    sc = scale;
    float nn = n * scale;
    sqn[row] = nn * nn;
  }
  __syncthreads();
  float scale = sc;
  for (int c = threadIdx.x; c < C; c += 256) {
    float v = f[c] * scale;
    f16[(size_t)row * C + c]     = (half_t)v;
    f16T[(size_t)c * 5120 + row] = (half_t)v;
    if (row < 1024) qf[(size_t)row * C + c] = v;
  }
}

// ------------------------- Gram matrix: out = A @ B^T ----------------------
// A: 1024 x K, B: N x K (f16 row-major), out: 1024 x N (f32).
// Each wave computes a 16x64 slab: one A-fragment feeds 4 WMMAs.
__global__ void gemm_nt_kernel(float* __restrict__ out, const half_t* __restrict__ A,
                               const half_t* __restrict__ B, int N, int K) {
  const int mt = blockIdx.x;
  const int nb = blockIdx.y * 64;
  const int lane = threadIdx.x;
  const int nloc = lane & 15, hi = lane >> 4;
  const half_t* aRow  = A + (size_t)(mt * 16 + nloc) * K;
  const half_t* bRow0 = B + (size_t)(nb + nloc) * K;
  const half_t* bRow1 = B + (size_t)(nb + 16 + nloc) * K;
  const half_t* bRow2 = B + (size_t)(nb + 32 + nloc) * K;
  const half_t* bRow3 = B + (size_t)(nb + 48 + nloc) * K;
  v8f acc0 = {}, acc1 = {}, acc2 = {}, acc3 = {};
  for (int kb = 0; kb < K; kb += 32) {
    v16h a = load_afrag(aRow, kb, hi);
    acc0 = wmma_f16(a, load_bfragT(bRow0, kb, hi), acc0);
    acc1 = wmma_f16(a, load_bfragT(bRow1, kb, hi), acc1);
    acc2 = wmma_f16(a, load_bfragT(bRow2, kb, hi), acc2);
    acc3 = wmma_f16(a, load_bfragT(bRow3, kb, hi), acc3);
  }
#pragma unroll
  for (int r = 0; r < 8; ++r) {
    size_t o = (size_t)(mt * 16 + r + hi * 8) * N + nb + nloc;
    out[o]      = acc0[r];
    out[o + 16] = acc1[r];
    out[o + 32] = acc2[r];
    out[o + 48] = acc3[r];
  }
}

// -------------------- softmax over [pos(4096) | neg(1024)] -----------------
// Writes wp (f16) and NEGATED wn (f16) so the V GEMMs share one accumulator.
__global__ void softmax_row_kernel(const float* __restrict__ Gp, const float* __restrict__ Gn,
                                   const float* __restrict__ sqn, float inv,
                                   half_t* __restrict__ wp, half_t* __restrict__ wn,
                                   float* __restrict__ sp, float* __restrict__ sn) {
  __shared__ float red[256];
  const int row = blockIdx.x, t = threadIdx.x;
  const float qn = sqn[row];
  float lg[20];
  float mx = -3.402823e38f;
#pragma unroll
  for (int i = 0; i < 16; ++i) {                 // positives: idx = t + i*256 < 4096
    int j = t + i * 256;
    float d = qn + sqn[1024 + j] - 2.f * Gp[(size_t)row * 4096 + j];
    float l = -fmaxf(d, 0.f) * inv;
    lg[i] = l; mx = fmaxf(mx, l);
  }
#pragma unroll
  for (int i = 16; i < 20; ++i) {                // negatives (diag = -inf)
    int j = t + (i - 16) * 256;
    float l;
    if (j == row) l = -__builtin_inff();
    else {
      float d = qn + sqn[j] - 2.f * Gn[(size_t)row * 1024 + j];
      l = -fmaxf(d, 0.f) * inv;
    }
    lg[i] = l; mx = fmaxf(mx, l);
  }
  red[t] = mx; __syncthreads();
  for (int s = 128; s > 0; s >>= 1) {
    if (t < s) red[t] = fmaxf(red[t], red[t + s]);
    __syncthreads();
  }
  mx = red[0]; __syncthreads();

  float sumAll = 0.f, sumPos = 0.f;
#pragma unroll
  for (int i = 0; i < 20; ++i) {
    float e = __expf(lg[i] - mx);
    lg[i] = e; sumAll += e; if (i < 16) sumPos += e;
  }
  sumAll = block_reduce_sum(sumAll, red);
  sumPos = block_reduce_sum(sumPos, red);
  float isum = 1.f / sumAll;
#pragma unroll
  for (int i = 0; i < 16; ++i)
    wp[(size_t)row * 4096 + t + i * 256] = (half_t)(lg[i] * isum);
#pragma unroll
  for (int i = 16; i < 20; ++i)
    wn[(size_t)row * 1024 + t + (i - 16) * 256] = (half_t)(-lg[i] * isum);
  if (t == 0) { sp[row] = sumPos * isum; sn[row] = (sumAll - sumPos) * isum; }
}

// --------- V = wp@p + (-wn)@q - (sp - sn) * q  (fused, K-contiguous) -------
// f16T is C x 5120 (q columns 0..1023, p columns 1024..5119): B-fragments are
// contiguous 32-byte loads. Each wave computes a 32x16 slab.
__global__ void drift_gemm_kernel(float* __restrict__ V, const half_t* __restrict__ wp,
                                  const half_t* __restrict__ wn,
                                  const half_t* __restrict__ f16T,
                                  const float* __restrict__ qf,
                                  const float* __restrict__ sp, const float* __restrict__ sn,
                                  int C) {
  const int mb = blockIdx.x * 32;
  const int nt = blockIdx.y;
  const int lane = threadIdx.x;
  const int nloc = lane & 15, hi = lane >> 4;
  const int n = nt * 16 + nloc;
  const half_t* bCol = f16T + (size_t)n * 5120;   // qT at +0, pT at +1024
  v8f acc0 = {}, acc1 = {};
  const half_t* aRow0 = wp + (size_t)(mb + nloc) * 4096;
  const half_t* aRow1 = wp + (size_t)(mb + 16 + nloc) * 4096;
  for (int kb = 0; kb < 4096; kb += 32) {
    __builtin_prefetch(bCol + 1024 + kb + 64, 0, 0);
    v16h b = load_bfragT(bCol + 1024, kb, hi);    // p columns
    acc0 = wmma_f16(load_afrag(aRow0, kb, hi), b, acc0);
    acc1 = wmma_f16(load_afrag(aRow1, kb, hi), b, acc1);
  }
  const half_t* cRow0 = wn + (size_t)(mb + nloc) * 1024;        // pre-negated
  const half_t* cRow1 = wn + (size_t)(mb + 16 + nloc) * 1024;
  for (int kb = 0; kb < 1024; kb += 32) {
    v16h b = load_bfragT(bCol, kb, hi);           // q columns
    acc0 = wmma_f16(load_afrag(cRow0, kb, hi), b, acc0);
    acc1 = wmma_f16(load_afrag(cRow1, kb, hi), b, acc1);
  }
#pragma unroll
  for (int r = 0; r < 8; ++r) {
    int m0 = mb + r + hi * 8;
    int m1 = m0 + 16;
    V[(size_t)m0 * C + n] = acc0[r] - (sp[m0] - sn[m0]) * qf[(size_t)m0 * C + n];
    V[(size_t)m1 * C + n] = acc1[r] - (sp[m1] - sn[m1]) * qf[(size_t)m1 * C + n];
  }
}

// ---------------------- drift normalization statistics ---------------------
__global__ void row_stats_kernel(const float* __restrict__ V, int C, float* __restrict__ stats) {
  __shared__ float red[256];
  const int row = blockIdx.x;
  float partial = 0.f;
  for (int c = threadIdx.x; c < C; c += 256) {
    float v = V[(size_t)row * C + c];
    partial += v * v;
  }
  float ss = block_reduce_sum(partial, red);
  if (threadIdx.x == 0) {
    atomicAdd(&stats[0], sqrtf(ss) + 1e-8f);   // norms = ||V_i|| + EPS
    atomicAdd(&stats[1], ss);                  // total sum of squares
  }
}

__global__ void finish_kernel(const float* __restrict__ stats, int C, float* __restrict__ out) {
  if (threadIdx.x == 0 && blockIdx.x == 0) {
    float denom = stats[0] * (1.f / 1024.f) + 1e-8f;     // norms.mean() + EPS
    float loss  = stats[1] / (denom * denom) * (1.f / (1024.f * (float)C)) * 0.25f;
    atomicAdd(out, loss);
  }
}

// ------------------------------- host driver -------------------------------

extern "C" void kernel_launch(void* const* d_in, const int* in_sizes, int n_in,
                              void* d_out, int out_size, void* d_ws, size_t ws_size,
                              hipStream_t stream) {
  (void)in_sizes; (void)n_in; (void)out_size; (void)ws_size;

  const float* x_gen  = (const float*)d_in[0];
  const float* x_data = (const float*)d_in[1];
  const float* W[4]  = {(const float*)d_in[2], (const float*)d_in[4],
                        (const float*)d_in[6], (const float*)d_in[8]};
  const float* Bw[4] = {(const float*)d_in[3], (const float*)d_in[5],
                        (const float*)d_in[7], (const float*)d_in[9]};
  float* out = (float*)d_out;

  // Workspace bump allocator (256-byte aligned blocks)
  char* base = (char*)d_ws;
  size_t off = 0;
  auto alloc = [&](size_t bytes) -> char* {
    char* p = base + off;
    off = (off + bytes + 255) & ~(size_t)255;
    return p;
  };

  half_t* bufA = (half_t*)alloc((size_t)8388608 * 2);   // chunk act ping
  half_t* bufB = (half_t*)alloc((size_t)8388608 * 2);   // chunk act pong
  struct LayerP { int Cin, Hi, Cout, Ho, Kr, Kpad; float invS; int colofs; };
  const LayerP L[4] = {
      {4,   32, 64,  16, 36,   64,   1.f / 256.f, 0},
      {64,  16, 128, 8,  576,  576,  1.f / 64.f,  64},
      {128, 8,  256, 4,  1152, 1152, 1.f / 16.f,  192},
      {256, 4,  512, 2,  2304, 2304, 1.f / 4.f,   448}};
  half_t* wk[4];
  int*    tab[4];
  for (int l = 0; l < 4; ++l) {
    wk[l]  = (half_t*)alloc((size_t)L[l].Cout * L[l].Kpad * 2);
    tab[l] = (int*)alloc((size_t)L[l].Kpad * 4);
  }

  float*  pooled = (float*)alloc((size_t)5120 * 960 * 4);  // pooled means
  half_t* fbuf   = (half_t*)alloc((size_t)5120 * 512 * 2); // normalized feats (row-major)
  half_t* f16T   = (half_t*)alloc((size_t)512 * 5120 * 2); // normalized feats (C x 5120)
  float*  qf     = (float*)alloc((size_t)1024 * 512 * 4);  // normalized q f32
  float*  sqn    = (float*)alloc((size_t)5120 * 4);        // squared norms
  float*  Gpos   = (float*)alloc((size_t)1024 * 4096 * 4);
  float*  Gneg   = (float*)alloc((size_t)1024 * 1024 * 4);
  half_t* wp     = (half_t*)alloc((size_t)1024 * 4096 * 2);
  half_t* wn     = (half_t*)alloc((size_t)1024 * 1024 * 2);
  float*  sp     = (float*)alloc((size_t)1024 * 4);
  float*  sn     = (float*)alloc((size_t)1024 * 4);
  float*  V      = (float*)alloc((size_t)1024 * 512 * 4);
  float*  stats  = (float*)alloc(2 * 4);

  // Init accumulators
  zero_f32_kernel<<<1, 32, 0, stream>>>(out, 1);
  zero_f32_kernel<<<(5120 * 960 + 255) / 256, 256, 0, stream>>>(pooled, 5120 * 960);

  // Weight prep (f32 OIHW -> f16 im2col rows + gather tables)
  for (int l = 0; l < 4; ++l) {
    int n = L[l].Cout * L[l].Kpad;
    prep_weights_kernel<<<(n + 255) / 256, 256, 0, stream>>>(
        wk[l], tab[l], W[l], L[l].Cout, L[l].Kr, L[l].Kpad, L[l].Hi);
  }

  // Feature extraction: 10 chunks of 512 images (0..1023 gen, 1024..5119 data)
  for (int ch = 0; ch < 10; ++ch) {
    int imgBase = ch * 512;
    const float* src = (ch < 2) ? x_gen + (size_t)ch * 512 * 4096
                                : x_data + (size_t)(ch - 2) * 512 * 4096;
    cvt_f16_kernel<<<8192, 256, 0, stream>>>(bufA, src, 512 * 4096);
    half_t* pin = bufA;
    half_t* pout = bufB;
    for (int l = 0; l < 4; ++l) {
      int N = L[l].Ho * L[l].Ho;
      int ntiles = (N + 15) / 16;
      int mtiles = L[l].Cout / 16;
      conv_wmma_kernel<<<dim3(512), 256, 0, stream>>>(
          pin, pout, wk[l], tab[l], Bw[l], pooled, L[l].Hi, L[l].Cout, L[l].Ho,
          L[l].Cin * L[l].Hi * L[l].Hi, L[l].Kpad, L[l].invS, imgBase, L[l].colofs,
          mtiles, ntiles);
      half_t* t = pin; pin = pout; pout = t;
    }
  }

  // Attention / drift loss per scale
  const int   Cs[4]    = {64, 128, 256, 512};
  const float temps[4] = {0.1f, 0.5f, 1.f, 2.f};
  for (int s = 0; s < 4; ++s) {
    int C = Cs[s];
    norm_pool_kernel<<<5120, 256, 0, stream>>>(pooled, L[s].colofs, C, fbuf, f16T, qf, sqn);
    // Gram matrices, computed ONCE per scale, reused across all 4 temperatures
    gemm_nt_kernel<<<dim3(64, 64), 32, 0, stream>>>(Gpos, fbuf,
                                                    fbuf + (size_t)1024 * C, 4096, C);
    gemm_nt_kernel<<<dim3(64, 16), 32, 0, stream>>>(Gneg, fbuf, fbuf, 1024, C);
    for (int ti = 0; ti < 4; ++ti) {
      float inv = sqrtf((float)C) / (2.f * temps[ti]);  // 1/(2*sigma^2*temp/sqrt(C))
      softmax_row_kernel<<<1024, 256, 0, stream>>>(Gpos, Gneg, sqn, inv, wp, wn, sp, sn);
      drift_gemm_kernel<<<dim3(32, C / 16), 32, 0, stream>>>(V, wp, wn, f16T, qf, sp, sn, C);
      zero_f32_kernel<<<1, 32, 0, stream>>>(stats, 2);
      row_stats_kernel<<<1024, 256, 0, stream>>>(V, C, stats);
      finish_kernel<<<1, 1, 0, stream>>>(stats, C, out);
    }
  }
}
